// crossAttention_8083128451560
// MI455X (gfx1250) — compile-verified, matrix operand order checked by reading, not run on hardware
//
#include <hip/hip_runtime.h>
#include <hip/hip_bf16.h>
#include <math.h>

// ---------------------------------------------------------------------------
// CDNA5 (gfx1250, wave32) cross-attention:
//   Q=x@Wq+bq; K=ctx@Wk+bk; Vt=(ctx@Wv+bv)^T; S=QK^T/sqrt(D); P=softmax(S); O=P@V
// Matmuls: v_wmma_f32_16x16x32_bf16.  Attention GEMM tiles staged by the
// Tensor Data Mover (tensor_load_to_lds, TENSORcnt) with double buffering.
// ---------------------------------------------------------------------------

typedef __attribute__((ext_vector_type(16))) __bf16   v16bf;
typedef __attribute__((ext_vector_type(8)))  __bf16   v8bf;
typedef __attribute__((ext_vector_type(8)))  float    v8f;
typedef __attribute__((ext_vector_type(4)))  unsigned u32x4;
typedef __attribute__((ext_vector_type(8)))  int      i32x8;
typedef __attribute__((ext_vector_type(4)))  int      i32x4;

#define LDS_STRIDE 40   // bf16 elems per LDS row: 80B = 20 banks -> conflict-free
#define TILE_ELEMS (128 * LDS_STRIDE)

#if defined(__has_builtin)
#  if __has_builtin(__builtin_amdgcn_tensor_load_to_lds)
#    define HAVE_TDM 1
#  endif
#endif
#ifndef HAVE_TDM
#  define HAVE_TDM 0
#endif

#if HAVE_TDM
// Issue one TDM 2D tile load: 128 rows x 32 bf16, row stride ld_elems in global,
// LDS rows padded to 40 elems (pad 4 DWORDs after every 16 DWORDs stored).
static __device__ inline void tdm_tile_load(unsigned lds_off, const void* gp, int ld_elems) {
    unsigned long long ga = (unsigned long long)(size_t)gp;
    u32x4 g0;
    g0[0] = 1u;                                            // count=1 (valid), user mode
    g0[1] = lds_off;                                       // lds_addr (bytes)
    g0[2] = (unsigned)(ga & 0xFFFFFFFFull);                // global_addr[31:0]
    g0[3] = (unsigned)((ga >> 32) & 0x1FFFFFFull)          // global_addr[56:32]
          | (2u << 30);                                    // type = 2 ("image")
    i32x8 g1;
    g1[0] = (1 << 16)      // data_size = 1 -> 2 bytes
          | (1 << 20)      // pad_enable
          | (3 << 22)      // pad_interval = 3 -> every 16 DWORDs (one 64B row)
          | (3 << 25);     // pad_amount   = 3 -> 4 DWORDs (8 bf16) -> stride 40
    g1[1] = 0;                                             // atomic_barrier_addr=0, TD0 lo16=0
    g1[2] = 16;                                            // tensor_dim0 = 1<<20 (hi16), TD1 lo16=0
    g1[3] = 16 | (32 << 16);                               // tensor_dim1 = 1<<20 (hi16), tile_dim0=32
    g1[4] = 128;                                           // tile_dim1 = 128, tile_dim2 = 0
    g1[5] = ld_elems;                                      // tensor_dim0_stride[31:0]
    g1[6] = 0;                                             // stride hi, dim1_stride lo
    g1[7] = 0;
    i32x4 z4 = {0, 0, 0, 0};
#if defined(__clang_major__) && (__clang_major__ >= 23)
    i32x8 z8 = {0, 0, 0, 0, 0, 0, 0, 0};
    __builtin_amdgcn_tensor_load_to_lds(g0, g1, z4, z4, z8, 0);
#else
    __builtin_amdgcn_tensor_load_to_lds(g0, g1, z4, z4, 0);
#endif
}
#endif

// Per-lane 16x32 bf16 WMMA A/B fragment from an LDS row (ISA 16-bit A layout):
// lanes 0-15: 8 bf16 at K[kbase..], 8 at K[kbase+16..]; lanes 16-31: kbase+8.
static __device__ inline v16bf load_frag(const __bf16* rowp, int kbase) {
    v8bf lo = *reinterpret_cast<const v8bf*>(rowp + kbase);
    v8bf hi = *reinterpret_cast<const v8bf*>(rowp + kbase + 16);
    return __builtin_shufflevector(lo, hi, 0,1,2,3,4,5,6,7,8,9,10,11,12,13,14,15);
}

// TDM==false: A is f32 [M][K], B is f32 [K][N] (weights, transposed into LDS).
// TDM==true : A is bf16 [M][K], B is bf16 [N][K]; both tiles loaded by TDM.
// OUT_MODE: 0 = f32 C[M][N], 1 = bf16 C[M][N], 2 = bf16 transposed C[N][M].
template <bool TDM, bool BIAS, int OUT_MODE>
__global__ __launch_bounds__(256)
void gemm_wmma_bf16(const void* __restrict__ Aop, const void* __restrict__ Bop,
                    const float* __restrict__ bias, void* __restrict__ Cop,
                    int M, int N, int K, int lda, int ldb, int ldc,
                    long long sAb, long long sBb, long long sCb, float scale) {
    constexpr int NB = TDM ? 2 : 1;          // double-buffer only the TDM path
    __shared__ __bf16 lsA[NB * TILE_ELEMS];
    __shared__ __bf16 lsB[NB * TILE_ELEMS];

    const int tid   = threadIdx.x;
    const int lane  = tid & 31;
    const int wid   = tid >> 5;              // 8 waves
    const int m0    = (wid & 3) * 32;        // 4x2 wave grid, wave tile 32x64
    const int n0w   = (wid >> 2) * 64;

    const int blockM = blockIdx.y * 128;
    const int blockN = blockIdx.x * 128;
    const size_t zA = (size_t)blockIdx.z * (size_t)sAb;
    const size_t zB = (size_t)blockIdx.z * (size_t)sBb;
    const size_t zC = (size_t)blockIdx.z * (size_t)sCb;

    const int r    = lane & 15;
    const int hi16 = lane >> 4;
    const int kb8  = hi16 << 3;

    v8f acc[2][4];
    #pragma unroll
    for (int mi = 0; mi < 2; ++mi)
        #pragma unroll
        for (int ni = 0; ni < 4; ++ni)
            acc[mi][ni] = v8f{0.f,0.f,0.f,0.f,0.f,0.f,0.f,0.f};

    const int nIter = K >> 5;

    if constexpr (TDM) {
        const __bf16* Abf = (const __bf16*)Aop + zA + (size_t)blockM * lda;
        const __bf16* Bbf = (const __bf16*)Bop + zB + (size_t)blockN * ldb;
#if HAVE_TDM
        const unsigned ldsA0 = (unsigned)(size_t)(void*)&lsA[0];
        const unsigned ldsB0 = (unsigned)(size_t)(void*)&lsB[0];
        if (wid == 0) {                       // prologue: fill buffer 0
            tdm_tile_load(ldsA0, Abf, lda);
            tdm_tile_load(ldsB0, Bbf, ldb);
        }
#endif
        for (int it = 0; it < nIter; ++it) {
            const int cur    = it & 1;
            const int curOff = cur * TILE_ELEMS;
#if HAVE_TDM
            if (wid == 0) {
                if (it + 1 < nIter) {         // prefetch next tile into other buffer
                    const unsigned nxtB = (unsigned)((cur ^ 1) * TILE_ELEMS * 2);
                    tdm_tile_load(ldsA0 + nxtB, Abf + (size_t)(it + 1) * 32, lda);
                    tdm_tile_load(ldsB0 + nxtB, Bbf + (size_t)(it + 1) * 32, ldb);
                    __builtin_amdgcn_s_wait_tensorcnt(2);  // current tile done
                } else {
                    __builtin_amdgcn_s_wait_tensorcnt(0);
                }
            }
#else
            {   // fallback: cooperative direct bf16 copy
                const int row = tid >> 1;
                const int c0  = (tid & 1) << 4;
                const __bf16* pa = Abf + (size_t)row * lda + (size_t)it * 32 + c0;
                const __bf16* pb = Bbf + (size_t)row * ldb + (size_t)it * 32 + c0;
                __bf16* da = &lsA[curOff + row * LDS_STRIDE + c0];
                __bf16* db = &lsB[curOff + row * LDS_STRIDE + c0];
                *reinterpret_cast<v8bf*>(da)     = *reinterpret_cast<const v8bf*>(pa);
                *reinterpret_cast<v8bf*>(da + 8) = *reinterpret_cast<const v8bf*>(pa + 8);
                *reinterpret_cast<v8bf*>(db)     = *reinterpret_cast<const v8bf*>(pb);
                *reinterpret_cast<v8bf*>(db + 8) = *reinterpret_cast<const v8bf*>(pb + 8);
            }
#endif
            __syncthreads();                  // tile ready for all waves

            v16bf afrag[2], bfrag[4];
            #pragma unroll
            for (int mi = 0; mi < 2; ++mi)
                afrag[mi] = load_frag(&lsA[curOff + (m0 + mi*16 + r) * LDS_STRIDE], kb8);
            #pragma unroll
            for (int ni = 0; ni < 4; ++ni)
                bfrag[ni] = load_frag(&lsB[curOff + (n0w + ni*16 + r) * LDS_STRIDE], kb8);
            #pragma unroll
            for (int mi = 0; mi < 2; ++mi)
                #pragma unroll
                for (int ni = 0; ni < 4; ++ni)
                    acc[mi][ni] = __builtin_amdgcn_wmma_f32_16x16x32_bf16(
                        false, afrag[mi], false, bfrag[ni],
                        (short)0, acc[mi][ni], false, false);
            __syncthreads();                  // readers done before buffer reuse
        }
    } else {
        // f32 A + f32 [K][N] weights, converted to bf16 while staging to LDS.
        for (int it = 0; it < nIter; ++it) {
            const int kb = it * 32;
            {   // A tile: 2 threads/row, 16 f32 each -> cvt -> 2x 16B LDS stores
                const int row  = tid >> 1;
                const int col0 = (tid & 1) << 4;
                const float* pa = (const float*)Aop + zA +
                                  (size_t)(blockM + row) * lda + (kb + col0);
                v8bf lo, hi;
                #pragma unroll
                for (int j = 0; j < 8; ++j) lo[j] = (__bf16)pa[j];
                #pragma unroll
                for (int j = 0; j < 8; ++j) hi[j] = (__bf16)pa[8 + j];
                __bf16* dst = &lsA[row * LDS_STRIDE + col0];
                *reinterpret_cast<v8bf*>(dst)     = lo;
                *reinterpret_cast<v8bf*>(dst + 8) = hi;
            }
            {   // B tile: transpose [k][n] -> LDS [n][k]
                const int kk  = tid >> 3;
                const int n0l = (tid & 7) << 4;
                const float* pb = (const float*)Bop + zB +
                                  (size_t)(kb + kk) * ldb + (blockN + n0l);
                #pragma unroll
                for (int j = 0; j < 16; ++j)
                    lsB[(n0l + j) * LDS_STRIDE + kk] = (__bf16)pb[j];
            }
            __syncthreads();

            v16bf afrag[2], bfrag[4];
            #pragma unroll
            for (int mi = 0; mi < 2; ++mi)
                afrag[mi] = load_frag(&lsA[(m0 + mi*16 + r) * LDS_STRIDE], kb8);
            #pragma unroll
            for (int ni = 0; ni < 4; ++ni)
                bfrag[ni] = load_frag(&lsB[(n0w + ni*16 + r) * LDS_STRIDE], kb8);
            #pragma unroll
            for (int mi = 0; mi < 2; ++mi)
                #pragma unroll
                for (int ni = 0; ni < 4; ++ni)
                    acc[mi][ni] = __builtin_amdgcn_wmma_f32_16x16x32_bf16(
                        false, afrag[mi], false, bfrag[ni],
                        (short)0, acc[mi][ni], false, false);
            __syncthreads();
        }
    }

    // ---- epilogue ----
    #pragma unroll
    for (int mi = 0; mi < 2; ++mi) {
        #pragma unroll
        for (int ni = 0; ni < 4; ++ni) {
            const int gmBase = blockM + m0 + mi * 16 + (hi16 << 3);
            const int gn     = blockN + n0w + ni * 16 + r;
            float badd = 0.f;
            if constexpr (BIAS) badd = bias[gn];
            if constexpr (OUT_MODE == 2) {
                v8bf o;                       // transposed: one 16B store per lane
                #pragma unroll
                for (int v = 0; v < 8; ++v)
                    o[v] = (__bf16)(acc[mi][ni][v] * scale + badd);
                *reinterpret_cast<v8bf*>((__bf16*)Cop + zC + (size_t)gn * ldc + gmBase) = o;
            } else {
                #pragma unroll
                for (int v = 0; v < 8; ++v) {
                    const float  val = acc[mi][ni][v] * scale + badd;
                    const size_t idx = zC + (size_t)(gmBase + v) * ldc + gn;
                    if constexpr (OUT_MODE == 1) ((__bf16*)Cop)[idx] = (__bf16)val;
                    else                         ((float*)Cop)[idx]  = val;
                }
            }
        }
    }
}

// ---------------------------------------------------------------------------
// Row softmax over bf16 scores, in place. One 256-thread block per row of 2048.
// ---------------------------------------------------------------------------
__global__ __launch_bounds__(256)
void softmax_rows_bf16(__bf16* __restrict__ S, int ncols) {
    __bf16* row = S + (size_t)blockIdx.x * (size_t)ncols;
    const int tid  = threadIdx.x;
    const int lane = tid & 31;
    const int wid  = tid >> 5;

    float v[8];
    #pragma unroll
    for (int j = 0; j < 8; ++j) v[j] = (float)row[tid + j * 256];

    __shared__ float redmax[8];
    __shared__ float redsum[8];

    float m = -INFINITY;
    #pragma unroll
    for (int j = 0; j < 8; ++j) m = fmaxf(m, v[j]);
    #pragma unroll
    for (int off = 16; off > 0; off >>= 1) m = fmaxf(m, __shfl_xor(m, off));
    if (lane == 0) redmax[wid] = m;
    __syncthreads();
    m = -INFINITY;
    #pragma unroll
    for (int i = 0; i < 8; ++i) m = fmaxf(m, redmax[i]);

    float s = 0.f;
    #pragma unroll
    for (int j = 0; j < 8; ++j) { v[j] = expf(v[j] - m); s += v[j]; }
    #pragma unroll
    for (int off = 16; off > 0; off >>= 1) s += __shfl_xor(s, off);
    if (lane == 0) redsum[wid] = s;
    __syncthreads();
    s = 0.f;
    #pragma unroll
    for (int i = 0; i < 8; ++i) s += redsum[i];

    const float inv = 1.f / s;
    #pragma unroll
    for (int j = 0; j < 8; ++j) row[tid + j * 256] = (__bf16)(v[j] * inv);
}

// ---------------------------------------------------------------------------
extern "C" void kernel_launch(void* const* d_in, const int* in_sizes, int n_in,
                              void* d_out, int out_size, void* d_ws, size_t ws_size,
                              hipStream_t stream) {
    (void)in_sizes; (void)n_in; (void)out_size; (void)ws_size;

    constexpr int D   = 1024;
    constexpr int B   = 4;
    constexpr int SQ  = 2048;
    constexpr int SKV = 2048;

    const float* x   = (const float*)d_in[0];
    const float* ctx = (const float*)d_in[1];
    const float* Wq  = (const float*)d_in[2];
    const float* bq  = (const float*)d_in[3];
    const float* Wk  = (const float*)d_in[4];
    const float* bk  = (const float*)d_in[5];
    const float* Wv  = (const float*)d_in[6];
    const float* bv  = (const float*)d_in[7];
    float* out = (float*)d_out;

    // workspace carve (bf16): Qb(16MB) Kb(16MB) Vt(16MB) P(32MB) => 80MB
    char* w = (char*)d_ws;
    __bf16* Qb = (__bf16*)w;  w += (size_t)B * SQ  * D * 2;
    __bf16* Kb = (__bf16*)w;  w += (size_t)B * SKV * D * 2;
    __bf16* Vt = (__bf16*)w;  w += (size_t)B * D * SKV * 2;   // [b][d][kv]
    __bf16* P  = (__bf16*)w;

    const dim3 blk(256);
    const float rsqD = 0.03125f;  // 1/sqrt(1024)

    // --- projections (f32 staging path): Q, K row-major; V stored transposed ---
    gemm_wmma_bf16<false, true, 1><<<dim3(D/128, (B*SQ)/128, 1), blk, 0, stream>>>(
        x, Wq, bq, Qb, B*SQ, D, D, D, D, D, 0LL, 0LL, 0LL, 1.0f);
    gemm_wmma_bf16<false, true, 1><<<dim3(D/128, (B*SKV)/128, 1), blk, 0, stream>>>(
        ctx, Wk, bk, Kb, B*SKV, D, D, D, D, D, 0LL, 0LL, 0LL, 1.0f);
    gemm_wmma_bf16<false, true, 2><<<dim3(D/128, SKV/128, B), blk, 0, stream>>>(
        ctx, Wv, bv, Vt, SKV, D, D, D, D, SKV,
        (long long)SKV * D, 0LL, (long long)D * SKV, 1.0f);

    // --- S = Q K^T / sqrt(D): TDM-staged bf16 GEMM (K rows feed B-fragments) ---
    gemm_wmma_bf16<true, false, 1><<<dim3(SKV/128, SQ/128, B), blk, 0, stream>>>(
        Qb, Kb, nullptr, P, SQ, SKV, D, D, D, SKV,
        (long long)SQ * D, (long long)SKV * D, (long long)SQ * SKV, rsqD);

    // --- softmax over kv, in place ---
    softmax_rows_bf16<<<B * SQ, 256, 0, stream>>>(P, SKV);

    // --- O = P V: TDM-staged bf16 GEMM, B = Vt [d][kv], f32 out ---
    gemm_wmma_bf16<true, false, 0><<<dim3(D/128, SQ/128, B), blk, 0, stream>>>(
        P, Vt, nullptr, out, SQ, D, SKV, SKV, SKV, D,
        (long long)SQ * SKV, (long long)D * SKV, (long long)SQ * D, 1.0f);
}
